// Attention_833223655524
// MI455X (gfx1250) — compile-verified, matrix operand order checked by reading, not run on hardware
//
#include <hip/hip_runtime.h>
#include <hip/hip_bf16.h>
#include <stdint.h>

// ---------------- problem constants (from reference setup_inputs) -----------
#define B_    2
#define S_    2048
#define D_    2048
#define QH_   16       // query heads
#define NKV_  4        // kv heads
#define HD_   128      // head dim
#define M_    (B_ * S_)                 // 4096 rows
#define NOUT_ ((QH_ + 2 * NKV_) * HD_)  // 3072 fused qkv cols
#define QHD_  (QH_ * HD_)               // 2048

typedef __attribute__((ext_vector_type(16))) __bf16 v16bf;
typedef __attribute__((ext_vector_type(8)))  float  v8f;

union FragBF { uint32_t u[8]; v16bf v; };

__device__ __forceinline__ unsigned short f2bf(float f) {
  union { float f; uint32_t u; } v; v.f = f;
  uint32_t u = v.u;
  return (unsigned short)((u + 0x7FFFu + ((u >> 16) & 1u)) >> 16);  // RNE
}

#define WMMA_BF16(A, Bf, C) \
  __builtin_amdgcn_wmma_f32_16x16x32_bf16(false, (A), false, (Bf), (short)0, (C), false, false)

// ---- CDNA5 async copy: global -> LDS, tracked with ASYNCcnt ----------------
__device__ __forceinline__ void async_load16(uint32_t lds_off,
                                             const unsigned short* gptr) {
  // GLOBAL_LOAD_ASYNC_TO_LDS_B128: VDST = LDS byte offset, VADDR = 64b address
  asm volatile("global_load_async_to_lds_b128 %0, %1, off"
               :: "v"(lds_off), "v"(gptr) : "memory");
}
__device__ __forceinline__ void wait_async4() {
#if __has_builtin(__builtin_amdgcn_s_wait_asynccnt)
  __builtin_amdgcn_s_wait_asynccnt(4);
#else
  asm volatile("s_wait_asynccnt 0x4" ::: "memory");
#endif
}
__device__ __forceinline__ void wait_async0() {
#if __has_builtin(__builtin_amdgcn_s_wait_asynccnt)
  __builtin_amdgcn_s_wait_asynccnt(0);
#else
  asm volatile("s_wait_asynccnt 0x0" ::: "memory");
#endif
}

// Load a 16x32 bf16 A-layout fragment from a row-major [*, ld] bf16 matrix.
// ISA 16-bit A layout: lanes 0-15 hold K 0..7 (v0-3) and 16..23 (v4-7);
// lanes 16-31 hold K 8..15 and 24..31. Two bf16 per dword, K contiguous.
// B fragments are identical loads from the transposed (N x K) matrix.
__device__ __forceinline__ v16bf load_frag(const unsigned short* base, int ld,
                                           int row0, int k0, int lane) {
  int r  = lane & 15;
  int hi = lane >> 4;
  const unsigned short* p = base + (size_t)(row0 + r) * ld + k0 + hi * 8;
  const uint32_t* a = (const uint32_t*)p;
  const uint32_t* b = (const uint32_t*)(p + 16);
  FragBF f;
  f.u[0] = a[0]; f.u[1] = a[1]; f.u[2] = a[2]; f.u[3] = a[3];
  f.u[4] = b[0]; f.u[5] = b[1]; f.u[6] = b[2]; f.u[7] = b[3];
  return f.v;
}

// Same fragment load from a padded LDS tile (row stride 40 elements = 80 B:
// 20 dwords/row -> 16 consecutive rows map to 16 distinct bank groups, and
// every 16-B sub-read stays 16-B aligned for ds_load_b128).
#define LDS_STRIDE 40
__device__ __forceinline__ v16bf lds_frag(const unsigned short* tile,
                                          int row0, int lane) {
  int r  = lane & 15;
  int hi = lane >> 4;
  const unsigned short* p = tile + (row0 + r) * LDS_STRIDE + hi * 8;
  const uint32_t* a = (const uint32_t*)p;
  const uint32_t* b = (const uint32_t*)(p + 16);
  FragBF f;
  f.u[0] = a[0]; f.u[1] = a[1]; f.u[2] = a[2]; f.u[3] = a[3];
  f.u[4] = b[0]; f.u[5] = b[1]; f.u[6] = b[2]; f.u[7] = b[3];
  return f.v;
}

// ---------------- conversion kernels ---------------------------------------
__global__ void cvt_x_kernel(const float* __restrict__ x,
                             unsigned short* __restrict__ xb, size_t n) {
  size_t i = (size_t)blockIdx.x * blockDim.x + threadIdx.x;
  if (i < n) xb[i] = f2bf(x[i]);
}

// Wt[nout][d] (row-major, K=D contiguous), nout = fused q/k/v output column.
__global__ void cvt_wqkv_kernel(const float* __restrict__ Wq,
                                const float* __restrict__ Wk,
                                const float* __restrict__ Wv,
                                unsigned short* __restrict__ Wt) {
  size_t idx = (size_t)blockIdx.x * blockDim.x + threadIdx.x;
  if (idx >= (size_t)NOUT_ * D_) return;
  int d    = (int)(idx & (D_ - 1));
  int nout = (int)(idx >> 11);
  float val;
  if (nout < QHD_) {
    int q = nout >> 7, h = nout & 127;
    val = Wq[((size_t)q * D_ + d) * HD_ + h];
  } else if (nout < QHD_ + NKV_ * HD_) {
    int n = (nout - QHD_) >> 7, h = nout & 127;
    val = Wk[((size_t)n * D_ + d) * HD_ + h];
  } else {
    int n = (nout - QHD_ - NKV_ * HD_) >> 7, h = nout & 127;
    val = Wv[((size_t)n * D_ + d) * HD_ + h];
  }
  Wt[(size_t)nout * D_ + d] = f2bf(val);
}

// WoT[d][q*H+h] = Wo[q][h][d]  (so output-proj B fragments load like A frags)
__global__ void cvt_wo_kernel(const float* __restrict__ Wo,
                              unsigned short* __restrict__ WoT) {
  size_t idx = (size_t)blockIdx.x * blockDim.x + threadIdx.x;
  if (idx >= (size_t)D_ * QHD_) return;
  int qh = (int)(idx & (QHD_ - 1));
  int d  = (int)(idx >> 11);
  WoT[(size_t)d * QHD_ + qh] = f2bf(Wo[(size_t)qh * D_ + d]);
}

// ---------------- LDS-staged bf16 WMMA GEMM ---------------------------------
// C(f32)[M,N] = A(bf16)[M,K] x Bt(bf16)[N,K]^T.
// 256 threads = 8 waves; block tile 128x128, K-step 32.
// A/B k-slices staged to LDS with GLOBAL_LOAD_ASYNC_TO_LDS_B128 (ASYNCcnt),
// double-buffered so the async copy overlaps the WMMA stream.
#define TILE_ELEMS (128 * LDS_STRIDE)   // 5120 elements = 10240 B
__global__ void gemm_bf16_kernel(const unsigned short* __restrict__ A,
                                 const unsigned short* __restrict__ Bt,
                                 float* __restrict__ C,
                                 int M, int N, int K) {
  __shared__ unsigned short lds[4 * TILE_ELEMS];  // 2 bufs x (A + B) = 40 KB

  int t     = threadIdx.x;
  int lane  = t & 31;
  int wslot = t >> 5;
  int wm    = wslot >> 1;            // 0..3 -> 32-row strip
  int wn    = wslot & 1;             // 0..1 -> 64-col strip
  int nblk  = N >> 7;
  int m0    = (blockIdx.x / nblk) << 7;
  int n0    = (blockIdx.x % nblk) << 7;

  // NOTE: compute LDS buffer pointers at runtime (no constant pointer arrays:
  // a const-initialized array of generic pointers to LDS becomes a global
  // with an addrspacecast static initializer, which ld.lld rejects).
  // Abuf(d) = lds + d*TILE; Bbuf(d) = lds + (2+d)*TILE.

  // Issue one stage: copy A[128x32] at (m0,k0) and Bt[128x32] at (n0,k0).
  // 512 16-byte chunks per matrix / 256 threads = 2 + 2 async loads/thread.
  auto issue_stage = [&](int k0, int buf) {
    unsigned short* Ab = lds + (size_t)buf * TILE_ELEMS;
    unsigned short* Bb = lds + (size_t)(2 + buf) * TILE_ELEMS;
    for (int c = 0; c < 2; c++) {
      int chunk = t * 2 + c;         // 0..511
      int row   = chunk >> 2;
      int col   = (chunk & 3) * 8;   // element offset (8 bf16 = 16 B)
      uint32_t aoff = (uint32_t)(uintptr_t)(Ab + row * LDS_STRIDE + col);
      async_load16(aoff, A + (size_t)(m0 + row) * K + k0 + col);
      uint32_t boff = (uint32_t)(uintptr_t)(Bb + row * LDS_STRIDE + col);
      async_load16(boff, Bt + (size_t)(n0 + row) * K + k0 + col);
    }
  };

  v8f acc[2][4];
  for (int i = 0; i < 2; i++)
    for (int j = 0; j < 4; j++)
      for (int e = 0; e < 8; e++) acc[i][j][e] = 0.0f;

  int nsteps = K >> 5;
  issue_stage(0, 0);
  for (int s = 0; s < nsteps; s++) {
    int buf = s & 1;
    if (s + 1 < nsteps) {
      issue_stage((s + 1) << 5, buf ^ 1);
      wait_async4();                 // stage s landed; s+1 still in flight
    } else {
      wait_async0();
    }
    __syncthreads();                 // all waves' async data visible

    const unsigned short* Ab = lds + (size_t)buf * TILE_ELEMS;
    const unsigned short* Bb = lds + (size_t)(2 + buf) * TILE_ELEMS;
    v16bf a0 = lds_frag(Ab, wm * 32,      lane);
    v16bf a1 = lds_frag(Ab, wm * 32 + 16, lane);
    v16bf b0 = lds_frag(Bb, wn * 64,      lane);
    v16bf b1 = lds_frag(Bb, wn * 64 + 16, lane);
    v16bf b2 = lds_frag(Bb, wn * 64 + 32, lane);
    v16bf b3 = lds_frag(Bb, wn * 64 + 48, lane);
    acc[0][0] = WMMA_BF16(a0, b0, acc[0][0]);
    acc[0][1] = WMMA_BF16(a0, b1, acc[0][1]);
    acc[0][2] = WMMA_BF16(a0, b2, acc[0][2]);
    acc[0][3] = WMMA_BF16(a0, b3, acc[0][3]);
    acc[1][0] = WMMA_BF16(a1, b0, acc[1][0]);
    acc[1][1] = WMMA_BF16(a1, b1, acc[1][1]);
    acc[1][2] = WMMA_BF16(a1, b2, acc[1][2]);
    acc[1][3] = WMMA_BF16(a1, b3, acc[1][3]);

    __syncthreads();                 // done reading buf before s+2 overwrites
  }

  int cl = lane & 15, hi = lane >> 4;
  for (int i = 0; i < 2; i++)
    for (int j = 0; j < 4; j++)
      for (int r = 0; r < 8; r++) {
        int row = m0 + wm * 32 + i * 16 + r + hi * 8;  // C: VGPR r -> M=r+8*half
        int col = n0 + wn * 64 + j * 16 + cl;
        C[(size_t)row * N + col] = acc[i][j][r];
      }
}

// ---------------- RoPE + layout kernels -------------------------------------
// qkv: [M, 3072] f32 fused.  Writes Qb [B,Q,S,H], Kb [B,N,S,H] bf16 (roped).
__global__ void rope_qk_kernel(const float* __restrict__ qkv,
                               const int* __restrict__ pos,
                               unsigned short* __restrict__ Qb,
                               unsigned short* __restrict__ Kb) {
  size_t idx = (size_t)blockIdx.x * blockDim.x + threadIdx.x;
  if (idx >= (size_t)M_ * (QH_ + NKV_) * 64) return;
  int i     = (int)(idx & 63);
  size_t t  = idx >> 6;
  int hd    = (int)(t % (QH_ + NKV_));
  int m     = (int)(t / (QH_ + NKV_));
  int b     = m >> 11;
  int s     = m & (S_ - 1);
  float fr  = __powf(10000.0f, -(2.0f * (float)i) / 128.0f);
  float ang = (float)pos[s] * fr;
  float sn, cs;
  __sincosf(ang, &sn, &cs);
  int colbase = (hd < QH_) ? hd * HD_ : QHD_ + (hd - QH_) * HD_;
  float x1 = qkv[(size_t)m * NOUT_ + colbase + i];
  float x2 = qkv[(size_t)m * NOUT_ + colbase + 64 + i];
  float o1 = x1 * cs - x2 * sn;
  float o2 = x2 * cs + x1 * sn;
  if (hd < QH_) {
    size_t o = ((size_t)(b * QH_ + hd) * S_ + s) * HD_;
    Qb[o + i] = f2bf(o1); Qb[o + 64 + i] = f2bf(o2);
  } else {
    int n = hd - QH_;
    size_t o = ((size_t)(b * NKV_ + n) * S_ + s) * HD_;
    Kb[o + i] = f2bf(o1); Kb[o + 64 + i] = f2bf(o2);
  }
}

// V stored transposed: Vt [B,N,H,S] bf16 so PV B-fragments are contiguous in T.
__global__ void v_trans_kernel(const float* __restrict__ qkv,
                               unsigned short* __restrict__ Vt) {
  size_t idx = (size_t)blockIdx.x * blockDim.x + threadIdx.x;
  if (idx >= (size_t)M_ * NKV_ * HD_) return;
  int h    = (int)(idx & 127);
  size_t t = idx >> 7;
  int n    = (int)(t & 3);
  int m    = (int)(t >> 2);
  int b    = m >> 11;
  int s    = m & (S_ - 1);
  float v  = qkv[(size_t)m * NOUT_ + QHD_ + NKV_ * HD_ + n * HD_ + h];
  Vt[((size_t)(b * NKV_ + n) * HD_ + h) * S_ + s] = f2bf(v);
}

// ---------------- flash attention (1 wave = 16 query rows of one head) ------
__global__ void flash_attn_kernel(const unsigned short* __restrict__ Qb,
                                  const unsigned short* __restrict__ Kb,
                                  const unsigned short* __restrict__ Vt,
                                  unsigned short* __restrict__ Aout) {
  __shared__ unsigned short Plds[8 * 16 * 32];   // per-wave 16x32 bf16 P tile
  int lane  = threadIdx.x & 31;
  int wslot = threadIdx.x >> 5;
  int wid   = blockIdx.x * 8 + wslot;
  int qblk  = wid & 127;          // S/16 = 128 query blocks
  int tt    = wid >> 7;
  int qh    = tt & 15;
  int b     = tt >> 4;
  int s0    = qblk << 4;
  int nkv   = qh >> 2;            // GQA: 16 q heads share 4 kv heads

  const unsigned short* Qh = Qb + (size_t)(b * QH_ + qh)  * S_ * HD_;
  const unsigned short* Kh = Kb + (size_t)(b * NKV_ + nkv) * S_ * HD_;
  const unsigned short* Vh = Vt + (size_t)(b * NKV_ + nkv) * HD_ * S_;
  unsigned short* Pw = &Plds[wslot * 16 * 32];

  // Q fragments for this 16-row block: 16x128 held in registers (4 frags)
  v16bf qf[4];
  for (int kk = 0; kk < 4; kk++) qf[kk] = load_frag(Qh, HD_, s0, kk * 32, lane);

  v8f oacc[8];                                   // 16x128 output accumulator
  for (int i = 0; i < 8; i++)
    for (int e = 0; e < 8; e++) oacc[i][e] = 0.0f;
  float mrow[8], lrow[8];
  for (int r = 0; r < 8; r++) { mrow[r] = -3.0e38f; lrow[r] = 0.0f; }

  int cl = lane & 15, hi = lane >> 4;
  const float scale = 0.08838834764831845f;      // 1/sqrt(128)

  for (int t0 = 0; t0 < s0 + 16; t0 += 32) {     // causal key-block loop
    v8f sc0, sc1;
    for (int e = 0; e < 8; e++) { sc0[e] = 0.0f; sc1[e] = 0.0f; }
    for (int kk = 0; kk < 4; kk++) {             // K-dim = H = 128
      v16bf b0 = load_frag(Kh, HD_, t0,      kk * 32, lane);
      v16bf b1 = load_frag(Kh, HD_, t0 + 16, kk * 32, lane);
      sc0 = WMMA_BF16(qf[kk], b0, sc0);
      sc1 = WMMA_BF16(qf[kk], b1, sc1);
    }
    // scale + causal mask + online softmax (rows live in 16-lane halves)
    for (int r = 0; r < 8; r++) {
      int row = s0 + r + 8 * hi;
      float a = sc0[r] * scale; if (t0 + cl      > row) a = -1.0e9f;
      float c = sc1[r] * scale; if (t0 + 16 + cl > row) c = -1.0e9f;
      float mx = fmaxf(a, c);
      for (int msk = 1; msk < 16; msk <<= 1) mx = fmaxf(mx, __shfl_xor(mx, msk, 32));
      float mnew  = fmaxf(mrow[r], mx);
      float alpha = __expf(mrow[r] - mnew);
      float p0 = __expf(a - mnew);
      float p1 = __expf(c - mnew);
      float ps = p0 + p1;
      for (int msk = 1; msk < 16; msk <<= 1) ps += __shfl_xor(ps, msk, 32);
      lrow[r] = lrow[r] * alpha + ps;
      mrow[r] = mnew;
      for (int i = 0; i < 8; i++) oacc[i][r] *= alpha;
      Pw[(r + 8 * hi) * 32 + cl]      = f2bf(p0);
      Pw[(r + 8 * hi) * 32 + 16 + cl] = f2bf(p1);
    }
    // cross-lane LDS hazard: ensure P stores are visible before A-frag reload
    asm volatile("s_wait_dscnt 0" ::: "memory");
    FragBF pf;
    {
      const unsigned short* pp = Pw + cl * 32 + hi * 8;  // A layout from LDS
      const uint32_t* a  = (const uint32_t*)pp;
      const uint32_t* bb = (const uint32_t*)(pp + 16);
      pf.u[0] = a[0];  pf.u[1] = a[1];  pf.u[2] = a[2];  pf.u[3] = a[3];
      pf.u[4] = bb[0]; pf.u[5] = bb[1]; pf.u[6] = bb[2]; pf.u[7] = bb[3];
    }
    for (int nt = 0; nt < 8; nt++) {             // O += P x V  (H in 8 tiles)
      v16bf vf = load_frag(Vh, S_, nt * 16, t0, lane);
      oacc[nt] = WMMA_BF16(pf.v, vf, oacc[nt]);
    }
    asm volatile("s_wait_dscnt 0" ::: "memory"); // WAR: reads done before next stores
  }

  // normalize and store bf16 attention output [B*S, Q*H]
  for (int r = 0; r < 8; r++) {
    float inv = 1.0f / lrow[r];
    int row = s0 + r + 8 * hi;
    size_t base = ((size_t)(b * S_ + row)) * QHD_ + (size_t)qh * HD_;
    for (int nt = 0; nt < 8; nt++)
      Aout[base + nt * 16 + cl] = f2bf(oacc[nt][r] * inv);
  }
}

// ---------------- launch -----------------------------------------------------
extern "C" void kernel_launch(void* const* d_in, const int* in_sizes, int n_in,
                              void* d_out, int out_size, void* d_ws, size_t ws_size,
                              hipStream_t stream) {
  const float* x   = (const float*)d_in[0];
  const int*   pos = (const int*)d_in[1];
  const float* Wq  = (const float*)d_in[2];
  const float* Wk  = (const float*)d_in[3];
  const float* Wv  = (const float*)d_in[4];
  const float* Wo  = (const float*)d_in[5];
  float* out = (float*)d_out;

  char* ws = (char*)d_ws;
  size_t off = 0;
  auto take = [&](size_t bytes) -> char* {
    char* p = ws + off;
    off += (bytes + 255) & ~(size_t)255;
    return p;
  };
  unsigned short* xb   = (unsigned short*)take((size_t)M_ * D_ * 2);
  unsigned short* Wt   = (unsigned short*)take((size_t)NOUT_ * D_ * 2);
  unsigned short* WoT  = (unsigned short*)take((size_t)D_ * QHD_ * 2);
  float*          qkv  = (float*)take((size_t)M_ * NOUT_ * 4);
  unsigned short* Qb   = (unsigned short*)take((size_t)B_ * QH_ * S_ * HD_ * 2);
  unsigned short* Kb   = (unsigned short*)take((size_t)B_ * NKV_ * S_ * HD_ * 2);
  unsigned short* Vt   = (unsigned short*)take((size_t)B_ * NKV_ * HD_ * S_ * 2);
  unsigned short* attn = (unsigned short*)take((size_t)M_ * QHD_ * 2);

  // 1. precision conversion / weight transposition (one-time, bandwidth-bound)
  {
    size_t n = (size_t)M_ * D_;
    cvt_x_kernel<<<(unsigned)((n + 255) / 256), 256, 0, stream>>>(x, xb, n);
  }
  {
    size_t n = (size_t)NOUT_ * D_;
    cvt_wqkv_kernel<<<(unsigned)((n + 255) / 256), 256, 0, stream>>>(Wq, Wk, Wv, Wt);
  }
  {
    size_t n = (size_t)D_ * QHD_;
    cvt_wo_kernel<<<(unsigned)((n + 255) / 256), 256, 0, stream>>>(Wo, WoT);
  }
  // 2. fused QKV projection: [4096,2048] x [2048,3072] -> qkv f32
  {
    int blocks = (M_ / 128) * (NOUT_ / 128);    // 768
    gemm_bf16_kernel<<<blocks, 256, 0, stream>>>(xb, Wt, qkv, M_, NOUT_, D_);
  }
  // 3. RoPE on q/k + v transpose, write bf16 attention inputs
  {
    size_t n = (size_t)M_ * (QH_ + NKV_) * 64;
    rope_qk_kernel<<<(unsigned)((n + 255) / 256), 256, 0, stream>>>(qkv, pos, Qb, Kb);
  }
  {
    size_t n = (size_t)M_ * NKV_ * HD_;
    v_trans_kernel<<<(unsigned)((n + 255) / 256), 256, 0, stream>>>(qkv, Vt);
  }
  // 4. causal flash attention: 4096 waves (B*Q*S/16), 8 waves per block
  flash_attn_kernel<<<(B_ * QH_ * (S_ / 16)) / 8, 256, 0, stream>>>(Qb, Kb, Vt, attn);
  // 5. output projection: [4096,2048] x [2048,2048] -> d_out f32
  {
    int blocks = (M_ / 128) * (D_ / 128);       // 512
    gemm_bf16_kernel<<<blocks, 256, 0, stream>>>(attn, WoT, out, M_, D_, QHD_);
  }
}
